// CausalMultiHeadAttention_72636486910613
// MI455X (gfx1250) — matrix operand order repaired and checked by reading, not compiled, not run on hardware
//
#include <hip/hip_runtime.h>
#include <hip/hip_bf16.h>

// ---------------------------------------------------------------------------
// Causal MHA with RoPE for gfx1250 (MI455X).  B=4, S=2048, D=1024, H=16, dk=64
// All GEMMs via v_wmma_f32_16x16x32_bf16; A-tiles staged to LDS with
// global_load_async_to_lds_b128 (ASYNCcnt); LDS offsets derived from the real
// &Ash address so the LDS global escapes (prevents undef-folding of ds loads);
// B fragments double-buffered to hide global latency behind wmma.
// ---------------------------------------------------------------------------

#define D_MODEL 1024
#define NHEAD   16
#define DK      64
#define SEQL    2048
#define BATCHN  4
#define BS      (BATCHN * SEQL)   // 8192 flat rows
#define APAD    8                 // LDS row pad (elems): rotates banks per row

typedef __bf16 bf16_t;
typedef bf16_t v16bf __attribute__((ext_vector_type(16)));
typedef bf16_t v8bf  __attribute__((ext_vector_type(8)));
typedef float  v8f   __attribute__((ext_vector_type(8)));

static __device__ inline v8f wmma_bf16(v16bf a, v16bf b, v8f c) {
    // (neg_a, A, neg_b, B, c_mod, C, reuse_a, reuse_b)
    return __builtin_amdgcn_wmma_f32_16x16x32_bf16(false, a, false, b,
                                                   (short)0, c, false, false);
}

static __device__ inline v16bf cat8(v8bf lo, v8bf hi) {
    v16bf r;
#pragma unroll
    for (int i = 0; i < 8; ++i) { r[i] = lo[i]; r[i + 8] = hi[i]; }
    return r;
}

// Stage one 16 x D_MODEL bf16 tile (row-major, padded rows) into LDS using
// CDNA5 async global->LDS copies.  256 threads, 8 x B128 per thread = 32KB.
// lds_off is computed from the actual &Ash address: low 32 bits of a generic
// LDS pointer are the LDS byte offset, and the ptrtoint escapes the global.
static __device__ inline void stage_tile_async(bf16_t (&Ash)[16][D_MODEL + APAD],
                                               const bf16_t* __restrict__ src_base,
                                               int tid) {
#pragma unroll
    for (int i = 0; i < 8; ++i) {
        const int chunk = i * 256 + tid;            // 0..2047 16-byte chunks
        const int row   = chunk >> 7;               // 0..15
        const int c8    = chunk & 127;              // 8-elem column group
        const unsigned lds_off = (unsigned)(size_t)&Ash[row][c8 * 8];
        const void* g = (const void*)(src_base + (long long)row * D_MODEL + c8 * 8);
        asm volatile("global_load_async_to_lds_b128 %0, %1, off"
                     :: "v"(lds_off), "v"(g) : "memory");
    }
    asm volatile("s_wait_asynccnt 0x0" ::: "memory");
    __syncthreads();
}

// ---------------------------------------------------------------------------
// Kernel 0: elementwise fp32 -> bf16 (one-shot; removes cvt from GEMM loops)
// ---------------------------------------------------------------------------
__global__ __launch_bounds__(256)
void cvt_bf16_kernel(const float* __restrict__ src, bf16_t* __restrict__ dst,
                     int n8) {
    const int i = blockIdx.x * 256 + threadIdx.x;
    if (i < n8) {
        const v8f v = *(const v8f*)(src + (long long)i * 8);
        *(v8bf*)(dst + (long long)i * 8) = __builtin_convertvector(v, v8bf);
    }
}

// ---------------------------------------------------------------------------
// Kernel 1: fused QKV projection + RoPE.
// grid = (BS/16, 2, 3), block = 256 (8 waves).  Block stages the 16xD x-tile
// to LDS once (async); wave w computes a 16x64 tile (head h = by*8+w) with
// 4 accumulators and double-buffered B fragments.
//   Qr,Kr : [B,H,S,DK] bf16 (RoPE applied)     Vt : [B,H,DK,S] bf16
// ---------------------------------------------------------------------------
__global__ __launch_bounds__(256)
void qkv_rope_kernel(const bf16_t* __restrict__ xb,
                     const long long* __restrict__ tpos,
                     const bf16_t* __restrict__ Wqb,
                     const bf16_t* __restrict__ Wkb,
                     const bf16_t* __restrict__ Wvb,
                     bf16_t* __restrict__ Qr,
                     bf16_t* __restrict__ Kr,
                     bf16_t* __restrict__ Vt) {
    __shared__ bf16_t Ash[16][D_MODEL + APAD];

    const int lane  = threadIdx.x & 31;
    const int wave  = threadIdx.x >> 5;
    const int half  = lane >> 4;
    const int lcol  = lane & 15;
    const int mtile = blockIdx.x;                     // 0..511
    const int h     = blockIdx.y * 8 + wave;          // head / 64-col group
    const int which = blockIdx.z;                     // 0:Q 1:K 2:V
    const bf16_t* __restrict__ W = (which == 0) ? Wqb : ((which == 1) ? Wkb : Wvb);

    stage_tile_async(Ash, xb + (long long)mtile * 16 * D_MODEL, threadIdx.x);

    // per-lane B-row base pointers (include half*16 K-offset of the B layout)
    const bf16_t* Wr[4];
#pragma unroll
    for (int n = 0; n < 4; ++n)
        Wr[n] = W + (long long)(h * 64 + n * 16 + lcol) * D_MODEL + half * 16;

    v8f acc[4];
#pragma unroll
    for (int n = 0; n < 4; ++n) acc[n] = (v8f){};

    // double-buffered B fragments
    v8bf cb0[4], cb1[4];
#pragma unroll
    for (int n = 0; n < 4; ++n) {
        cb0[n] = *(const v8bf*)(Wr[n]);
        cb1[n] = *(const v8bf*)(Wr[n] + 8);
    }

#pragma unroll 2
    for (int kc = 0; kc < D_MODEL / 32; ++kc) {
        const int kn = (kc + 1 < D_MODEL / 32) ? kc + 1 : kc;
        v8bf nb0[4], nb1[4];
#pragma unroll
        for (int n = 0; n < 4; ++n) {
            nb0[n] = *(const v8bf*)(Wr[n] + kn * 32);
            nb1[n] = *(const v8bf*)(Wr[n] + kn * 32 + 8);
        }

        // A fragment from LDS: row = lane%16, K(e)=half*16+e
        const v8bf a0 = *(const v8bf*)(&Ash[lcol][kc * 32 + half * 16]);
        const v8bf a1 = *(const v8bf*)(&Ash[lcol][kc * 32 + half * 16 + 8]);
        const v16bf a = cat8(a0, a1);

#pragma unroll
        for (int n = 0; n < 4; ++n)
            acc[n] = wmma_bf16(a, cat8(cb0[n], cb1[n]), acc[n]);

#pragma unroll
        for (int n = 0; n < 4; ++n) { cb0[n] = nb0[n]; cb1[n] = nb1[n]; }
    }

    // Epilogue: C row M = half*8+r (flat token), col = h*64 + n*16 + lcol
    if (which == 2) {
#pragma unroll
        for (int n = 0; n < 4; ++n) {
            const int dk = n * 16 + lcol;
#pragma unroll
            for (int r = 0; r < 8; ++r) {
                const int flat = mtile * 16 + half * 8 + r;
                const int b    = flat / SEQL;
                const int s    = flat - b * SEQL;
                Vt[(((long long)b * NHEAD + h) * DK + dk) * SEQL + s] =
                    (bf16_t)acc[n][r];
            }
        }
    } else {
        bf16_t* __restrict__ dst = (which == 0) ? Qr : Kr;
#pragma unroll
        for (int n = 0; n < 4; ++n) {
            const int   dk  = n * 16 + lcol;
            const int   p   = dk >> 1;
            const float inv = __expf(-(float)p * 0.28782313662425572f); // ln(1e4)/32
#pragma unroll
            for (int r = 0; r < 8; ++r) {
                const int   flat = mtile * 16 + half * 8 + r;
                const int   b    = flat / SEQL;
                const int   s    = flat - b * SEQL;
                const float pos  = (float)tpos[s];
                const float ang  = pos * inv;
                const float cs   = __cosf(ang);
                const float sn   = __sinf(ang);
                const float v    = acc[n][r];
                const float pv   = __shfl_xor(v, 1, 32);   // partner feature
                const float outv = (lcol & 1) ? (pv * sn + v * cs)
                                              : (v * cs - pv * sn);
                dst[(((long long)b * NHEAD + h) * SEQL + s) * DK + dk] =
                    (bf16_t)outv;
            }
        }
    }
}

// ---------------------------------------------------------------------------
// Kernel 2: causal flash attention.  grid = (B*H, S/64), block = 128 (4 waves)
// Wave handles one 16-row q-tile; kv processed in chunks of 32.
// ---------------------------------------------------------------------------
__global__ __launch_bounds__(128)
void flash_attn_kernel(const bf16_t* __restrict__ Qr,
                       const bf16_t* __restrict__ Kr,
                       const bf16_t* __restrict__ Vt,
                       bf16_t* __restrict__ AO) {
    __shared__ bf16_t Psh[4][16][32];             // per-wave P staging

    const int lane = threadIdx.x & 31;
    const int wave = threadIdx.x >> 5;
    const int half = lane >> 4;
    const int lcol = lane & 15;
    const int bh   = blockIdx.x;                  // 0..63
    const int qt   = blockIdx.y * 4 + wave;       // 0..127
    const int q0   = qt * 16;

    const long long baseK = (long long)bh * SEQL * DK;
    const long long baseV = (long long)bh * DK * SEQL;

    // Q A-fragments for the two 32-deep d-chunks (row = lane%16 = q0+lcol)
    v16bf aQ[2];
#pragma unroll
    for (int kc = 0; kc < 2; ++kc) {
        const long long qoff = baseK + (long long)(q0 + lcol) * DK + kc * 32 + half * 16;
        const v8bf qa = *(const v8bf*)(Qr + qoff);
        const v8bf qb = *(const v8bf*)(Qr + qoff + 8);
        aQ[kc] = cat8(qa, qb);
    }

    float m[8], l[8];
    v8f   O[4];
#pragma unroll
    for (int r = 0; r < 8; ++r) { m[r] = -1e30f; l[r] = 0.0f; }
#pragma unroll
    for (int n = 0; n < 4; ++n) O[n] = (v8f){};

    const float scale = 0.125f;                   // 1/sqrt(64)

    for (int kb = 0; kb < q0 + 16; kb += 32) {
        // ---- scores: 16x32 = Q(16x64) · Kᵀ(64x32), two 16x16 C fragments
        v8f s0 = {}, s1 = {};
#pragma unroll
        for (int kc = 0; kc < 2; ++kc) {
            const v16bf bK0 = *(const v16bf*)(Kr + baseK + (long long)(kb + lcol) * DK
                                              + kc * 32 + half * 16);
            const v16bf bK1 = *(const v16bf*)(Kr + baseK + (long long)(kb + 16 + lcol) * DK
                                              + kc * 32 + half * 16);
            s0 = wmma_bf16(aQ[kc], bK0, s0);
            s1 = wmma_bf16(aQ[kc], bK1, s1);
        }

        // ---- causal mask + online softmax (row = half*8+r across 16 lanes)
#pragma unroll
        for (int r = 0; r < 8; ++r) {
            const int q   = q0 + half * 8 + r;
            const int kv0 = kb + lcol;
            const int kv1 = kb + 16 + lcol;
            float v0 = (kv0 <= q) ? s0[r] * scale : -1e30f;
            float v1 = (kv1 <= q) ? s1[r] * scale : -1e30f;

            float mx = fmaxf(v0, v1);
            mx = fmaxf(mx, __shfl_xor(mx, 1, 32));
            mx = fmaxf(mx, __shfl_xor(mx, 2, 32));
            mx = fmaxf(mx, __shfl_xor(mx, 4, 32));
            mx = fmaxf(mx, __shfl_xor(mx, 8, 32));

            const float mnew = fmaxf(m[r], mx);
            const float corr = __expf(m[r] - mnew);
            const float p0   = __expf(v0 - mnew);
            const float p1   = __expf(v1 - mnew);

            float rs = p0 + p1;
            rs += __shfl_xor(rs, 1, 32);
            rs += __shfl_xor(rs, 2, 32);
            rs += __shfl_xor(rs, 4, 32);
            rs += __shfl_xor(rs, 8, 32);

            l[r] = l[r] * corr + rs;
            m[r] = mnew;
            O[0][r] *= corr; O[1][r] *= corr; O[2][r] *= corr; O[3][r] *= corr;

            Psh[wave][half * 8 + r][lcol]      = (bf16_t)p0;
            Psh[wave][half * 8 + r][16 + lcol] = (bf16_t)p1;
        }

        // wave-local LDS store->load ordering (waves are divergent: no barrier)
        asm volatile("s_wait_dscnt 0x0" ::: "memory");

        // ---- P A-fragment: row = lane%16, K = half*16 + e
        v16bf aP;
        const bf16_t* prow = &Psh[wave][lcol][0];
#pragma unroll
        for (int i = 0; i < 8; ++i) {
            aP[i]     = prow[half * 16 + i];
            aP[i + 8] = prow[half * 16 + 8 + i];
        }

        // ---- O += P(16x32) · V(32x64): 4 d-subtiles, B from transposed Vt
#pragma unroll
        for (int n = 0; n < 4; ++n) {
            const v16bf bV = *(const v16bf*)(Vt + baseV
                                             + (long long)(n * 16 + lcol) * SEQL
                                             + kb + half * 16);
            O[n] = wmma_bf16(aP, bV, O[n]);
        }
    }

    // ---- normalize & write attention output [B,S,D_MODEL] bf16
    const int b = bh >> 4;
    const int h = bh & 15;
#pragma unroll
    for (int r = 0; r < 8; ++r) {
        const int   q    = q0 + half * 8 + r;
        const float invl = 1.0f / l[r];
        const long long base = ((long long)(b * SEQL + q)) * D_MODEL + h * DK;
#pragma unroll
        for (int n = 0; n < 4; ++n)
            AO[base + n * 16 + lcol] = (bf16_t)(O[n][r] * invl);
    }
}

// ---------------------------------------------------------------------------
// Kernel 3: output projection  out = AO · Woᵀ  (f32 out)
// grid = (BS/16, 2), block = 256 (8 waves); A tile async-staged to LDS,
// B fragments double-buffered.
// ---------------------------------------------------------------------------
__global__ __launch_bounds__(256)
void out_proj_kernel(const bf16_t* __restrict__ AO,
                     const bf16_t* __restrict__ Wob,
                     float* __restrict__ out) {
    __shared__ bf16_t Ash[16][D_MODEL + APAD];

    const int lane  = threadIdx.x & 31;
    const int wave  = threadIdx.x >> 5;
    const int half  = lane >> 4;
    const int lcol  = lane & 15;
    const int mtile = blockIdx.x;
    const int g     = blockIdx.y * 8 + wave;          // 64-col group

    stage_tile_async(Ash, AO + (long long)mtile * 16 * D_MODEL, threadIdx.x);

    const bf16_t* Wr[4];
#pragma unroll
    for (int n = 0; n < 4; ++n)
        Wr[n] = Wob + (long long)(g * 64 + n * 16 + lcol) * D_MODEL + half * 16;

    v8f acc[4];
#pragma unroll
    for (int n = 0; n < 4; ++n) acc[n] = (v8f){};

    v8bf cb0[4], cb1[4];
#pragma unroll
    for (int n = 0; n < 4; ++n) {
        cb0[n] = *(const v8bf*)(Wr[n]);
        cb1[n] = *(const v8bf*)(Wr[n] + 8);
    }

#pragma unroll 2
    for (int kc = 0; kc < D_MODEL / 32; ++kc) {
        const int kn = (kc + 1 < D_MODEL / 32) ? kc + 1 : kc;
        v8bf nb0[4], nb1[4];
#pragma unroll
        for (int n = 0; n < 4; ++n) {
            nb0[n] = *(const v8bf*)(Wr[n] + kn * 32);
            nb1[n] = *(const v8bf*)(Wr[n] + kn * 32 + 8);
        }

        const v8bf a0 = *(const v8bf*)(&Ash[lcol][kc * 32 + half * 16]);
        const v8bf a1 = *(const v8bf*)(&Ash[lcol][kc * 32 + half * 16 + 8]);
        const v16bf a = cat8(a0, a1);

#pragma unroll
        for (int n = 0; n < 4; ++n)
            acc[n] = wmma_bf16(a, cat8(cb0[n], cb1[n]), acc[n]);

#pragma unroll
        for (int n = 0; n < 4; ++n) { cb0[n] = nb0[n]; cb1[n] = nb1[n]; }
    }

#pragma unroll
    for (int n = 0; n < 4; ++n)
#pragma unroll
        for (int r = 0; r < 8; ++r)
            out[(long long)(mtile * 16 + half * 8 + r) * D_MODEL
                + g * 64 + n * 16 + lcol] = acc[n][r];
}

// ---------------------------------------------------------------------------
extern "C" void kernel_launch(void* const* d_in, const int* in_sizes, int n_in,
                              void* d_out, int out_size, void* d_ws, size_t ws_size,
                              hipStream_t stream) {
    const float*     x    = (const float*)d_in[0];
    const long long* tpos = (const long long*)d_in[1];
    const float*     Wq   = (const float*)d_in[2];
    const float*     Wk   = (const float*)d_in[3];
    const float*     Wv   = (const float*)d_in[4];
    const float*     Wo   = (const float*)d_in[5];
    float*           out  = (float*)d_out;

    const size_t n_qkv = (size_t)BATCHN * NHEAD * SEQL * DK;   // 8M elems
    const size_t n_x   = (size_t)BS * D_MODEL;                 // 8M elems
    const size_t n_w   = (size_t)D_MODEL * D_MODEL;            // 1M elems

    char* ws = (char*)d_ws;
    bf16_t* Qr  = (bf16_t*)(ws);                     // 16 MB
    bf16_t* Kr  = (bf16_t*)(ws + 2 * n_qkv);         // 16 MB
    bf16_t* Vt  = (bf16_t*)(ws + 4 * n_qkv);         // 16 MB
    bf16_t* AO  = (bf16_t*)(ws + 6 * n_qkv);         // 16 MB
    bf16_t* xb  = (bf16_t*)(ws + 8 * n_qkv);         // 16 MB
    bf16_t* Wqb = (bf16_t*)(ws + 8 * n_qkv + 2 * n_x);
    bf16_t* Wkb = Wqb + n_w;
    bf16_t* Wvb = Wkb + n_w;
    bf16_t* Wob = Wvb + n_w;                         // +8 MB  (total 88 MB)

    // 0) one-shot fp32 -> bf16 conversions
    cvt_bf16_kernel<<<dim3((int)(n_x / 8 / 256)), 256, 0, stream>>>(x,  xb,  (int)(n_x / 8));
    cvt_bf16_kernel<<<dim3((int)(n_w / 8 / 256)), 256, 0, stream>>>(Wq, Wqb, (int)(n_w / 8));
    cvt_bf16_kernel<<<dim3((int)(n_w / 8 / 256)), 256, 0, stream>>>(Wk, Wkb, (int)(n_w / 8));
    cvt_bf16_kernel<<<dim3((int)(n_w / 8 / 256)), 256, 0, stream>>>(Wv, Wvb, (int)(n_w / 8));
    cvt_bf16_kernel<<<dim3((int)(n_w / 8 / 256)), 256, 0, stream>>>(Wo, Wob, (int)(n_w / 8));

    // 1) QKV projections + RoPE
    qkv_rope_kernel<<<dim3(BS / 16, 2, 3), 256, 0, stream>>>(
        xb, tpos, Wqb, Wkb, Wvb, Qr, Kr, Vt);

    // 2) causal flash attention
    flash_attn_kernel<<<dim3(BATCHN * NHEAD, SEQL / 64), 128, 0, stream>>>(
        Qr, Kr, Vt, AO);

    // 3) output projection
    out_proj_kernel<<<dim3(BS / 16, 2), 256, 0, stream>>>(AO, Wob, out);
}
